// LTX2Attention_1769526526453
// MI455X (gfx1250) — matrix operand order repaired and checked by reading, not compiled, not run on hardware
//
#include <hip/hip_runtime.h>
#include <hip/hip_bf16.h>

typedef __attribute__((ext_vector_type(16))) __bf16 v16bf;
typedef __attribute__((ext_vector_type(8)))  __bf16 v8bf;
typedef __attribute__((ext_vector_type(8)))  float  v8f;
typedef __attribute__((ext_vector_type(4)))  unsigned int u32x4;
typedef __attribute__((ext_vector_type(8)))  int i32x8;
typedef __attribute__((ext_vector_type(4)))  int i32x4;

#define HEADS  32
#define DH     64
#define SEQ    2048
#define DMODEL 2048
#define NTOK   4096   // B * SEQ
#define NKT    (SEQ / 64)

#ifndef USE_TDM
#define USE_TDM 1
#endif
#if !(defined(__has_builtin) && __has_builtin(__builtin_amdgcn_tensor_load_to_lds))
#undef USE_TDM
#define USE_TDM 0
#endif

// ---------------------------------------------------------------------------
// WMMA helpers (CDNA5: D = A(16x32 bf16) * B(32x16 bf16) + C(16x16 f32))
// ---------------------------------------------------------------------------
__device__ __forceinline__ v8f wmma_bf16(v16bf a, v16bf b, v8f c) {
  return __builtin_amdgcn_wmma_f32_16x16x32_bf16(false, a, false, b, (short)0, c,
                                                 false, false);
}

union FragU {
  v16bf v;
  struct { v8bf lo, hi; } s;
};

// A-matrix fragment (16x32) from row-major [m][k] LDS tile.
// lanes 0-15: M=lane, K={kOff+0..7, kOff+16..23};
// lanes 16-31: M=lane-16, K={kOff+8..15, kOff+24..31}.  Two 16B LDS loads.
__device__ __forceinline__ v16bf load_a_frag(const __bf16* lds, int stride,
                                             int rowBase, int kOff, int lane) {
  const int m  = rowBase + (lane & 15);
  const int kb = kOff + ((lane >> 4) << 3);
  const __bf16* p = lds + m * stride + kb;
  FragU u;
  u.s.lo = *reinterpret_cast<const v8bf*>(p);
  u.s.hi = *reinterpret_cast<const v8bf*>(p + 16);
  return u.v;
}

// B-matrix fragment (32x16) from a TRANSPOSED LDS tile Bt[n][k].
// lanes 0-15: N=lane, K={kOff+0..7, kOff+16..23};
// lanes 16-31: N=lane-16, K={kOff+8..15, kOff+24..31}.  Two 16B LDS loads.
__device__ __forceinline__ v16bf load_bt_frag(const __bf16* lds, int stride,
                                              int kOff, int colBase, int lane) {
  const int n  = colBase + (lane & 15);
  const int kh = kOff + ((lane >> 4) << 3);
  const __bf16* p = lds + n * stride + kh;
  FragU u;
  u.s.lo = *reinterpret_cast<const v8bf*>(p);
  u.s.hi = *reinterpret_cast<const v8bf*>(p + 16);
  return u.v;
}

#if USE_TDM
// ---------------------------------------------------------------------------
// Tensor Data Mover: 2D bf16 tile load -> LDS with row padding of 4 dwords
// after every 32 dwords (i.e. 64 bf16 row -> 72-element LDS stride).
// Issued by a single wave; EXEC ignored by hardware for TDM.
// ---------------------------------------------------------------------------
__device__ __forceinline__ void tdm_load_2d_bf16(
    unsigned ldsOff, const __bf16* gaddr, unsigned tensor_d0, unsigned tensor_d1,
    unsigned stride0, unsigned tile_d0, unsigned tile_d1) {
  const unsigned long long ga = (unsigned long long)(uintptr_t)gaddr;
  u32x4 g0;
  g0[0] = 1u;                                  // count=1 (valid user descriptor)
  g0[1] = ldsOff;                              // lds_addr (bytes)
  g0[2] = (unsigned)(ga & 0xffffffffull);      // global_addr[31:0]
  g0[3] = (unsigned)(ga >> 32) | (2u << 30);   // global_addr[56:32] | type=2
  i32x8 g1;
  // data_size=1 (2B), pad_enable=1, pad_interval=4 (32 dwords), pad_amount=3 (4 dwords)
  g1[0] = (int)((1u << 16) | (1u << 20) | (4u << 22) | (3u << 25));
  g1[1] = (int)((tensor_d0 & 0xffffu) << 16);                       // dim0[15:0]
  g1[2] = (int)((tensor_d0 >> 16) | ((tensor_d1 & 0xffffu) << 16)); // dim0[31:16]|dim1[15:0]
  g1[3] = (int)((tensor_d1 >> 16) | (tile_d0 << 16));               // dim1[31:16]|tile0
  g1[4] = (int)(tile_d1);                                           // tile1 (tile2=0)
  g1[5] = (int)stride0;                                             // dim0_stride[31:0]
  g1[6] = 0;
  g1[7] = 0;
  i32x4 z4 = {0, 0, 0, 0};
  i32x8 z8 = {0, 0, 0, 0, 0, 0, 0, 0};
  __builtin_amdgcn_tensor_load_to_lds(g0, g1, z4, z4, z8, 0);
}
#endif

// ---------------------------------------------------------------------------
// 1) fp32 -> bf16 convert
// ---------------------------------------------------------------------------
__global__ __launch_bounds__(256) void cvt_kernel(const float* __restrict__ in,
                                                  __bf16* __restrict__ out,
                                                  int n4) {
  int i = blockIdx.x * blockDim.x + threadIdx.x;
  if (i < n4) {
    float4 v = reinterpret_cast<const float4*>(in)[i];
    __bf16* o = out + (size_t)i * 4;
    o[0] = (__bf16)v.x; o[1] = (__bf16)v.y;
    o[2] = (__bf16)v.z; o[3] = (__bf16)v.w;
  }
}

// ---------------------------------------------------------------------------
// 2) GEMM: C[M,N] = A(bf16)[M,K] @ W(fp32->bf16)[K,N] + bias
//    Block tile 128x128, 8 waves (4x2), wave tile 32x64, K-tile 32.
//    W tile stored transposed in LDS (Bt[n][k]) for contiguous B fragments.
//    MODE 0: fp32 row-major out.  MODE 1: bf16 out in [B,H,Dh,S] layout
//    (packed 8-wide stores along S).
// ---------------------------------------------------------------------------
template <int MODE>
__global__ __launch_bounds__(256) void gemm_bf16_kernel(
    const __bf16* __restrict__ A, const float* __restrict__ W,
    const float* __restrict__ bias, float* __restrict__ outF,
    __bf16* __restrict__ outAttn, int M, int N, int K) {
  __shared__ alignas(16) __bf16 As[128][40];
  __shared__ alignas(16) __bf16 Bt[128][40];   // [n][k]

  const int t    = threadIdx.x;
  const int lane = t & 31;
  const int w    = t >> 5;
  const int wrow = w >> 1;  // 0..3  (rows of 32)
  const int wcol = w & 1;   // 0..1  (cols of 64)
  const int mBase = blockIdx.y * 128;
  const int nBase = blockIdx.x * 128;

  v8f acc[2][4] = {};

  const int ar = t >> 1, ac = (t & 1) * 16;   // A tile: 128 rows x 32 cols
  const int nn = (t & 31) * 4;                // W tile: 4 n-cols per thread
  const int k4 = (t >> 5) * 4;                // 4 k-rows per thread

  for (int kBase = 0; kBase < K; kBase += 32) {
    {  // A tile (bf16, 32B per thread)
      const uint4* g =
          reinterpret_cast<const uint4*>(A + (size_t)(mBase + ar) * K + kBase + ac);
      uint4* d = reinterpret_cast<uint4*>(&As[ar][ac]);
      d[0] = g[0]; d[1] = g[1];
    }
    {  // W tile: fp32 -> bf16, transposed store (pairs packed along k)
      float4 wv[4];
#pragma unroll
      for (int kk = 0; kk < 4; ++kk)
        wv[kk] = *reinterpret_cast<const float4*>(
            W + (size_t)(kBase + k4 + kk) * N + nBase + nn);
#pragma unroll
      for (int i = 0; i < 4; ++i) {
        union { __bf16 h[4]; uint2 u2; } pk;
        const float c0 = (i == 0) ? wv[0].x : (i == 1) ? wv[0].y : (i == 2) ? wv[0].z : wv[0].w;
        const float c1 = (i == 0) ? wv[1].x : (i == 1) ? wv[1].y : (i == 2) ? wv[1].z : wv[1].w;
        const float c2 = (i == 0) ? wv[2].x : (i == 1) ? wv[2].y : (i == 2) ? wv[2].z : wv[2].w;
        const float c3 = (i == 0) ? wv[3].x : (i == 1) ? wv[3].y : (i == 2) ? wv[3].z : wv[3].w;
        pk.h[0] = (__bf16)c0; pk.h[1] = (__bf16)c1;
        pk.h[2] = (__bf16)c2; pk.h[3] = (__bf16)c3;
        *reinterpret_cast<uint2*>(&Bt[nn + i][k4]) = pk.u2;
      }
    }
    if (kBase + 32 < K) {  // prefetch next tiles (global_prefetch_b8)
      __builtin_prefetch(A + (size_t)(mBase + ar) * K + kBase + 32 + ac, 0, 1);
      __builtin_prefetch(W + (size_t)(kBase + 32 + k4) * N + nBase + nn, 0, 1);
    }
    __syncthreads();

    v16bf af[2], bfr[4];
#pragma unroll
    for (int mi = 0; mi < 2; ++mi)
      af[mi] = load_a_frag(&As[0][0], 40, wrow * 32 + mi * 16, 0, lane);
#pragma unroll
    for (int nj = 0; nj < 4; ++nj)
      bfr[nj] = load_bt_frag(&Bt[0][0], 40, 0, wcol * 64 + nj * 16, lane);
#pragma unroll
    for (int mi = 0; mi < 2; ++mi)
#pragma unroll
      for (int nj = 0; nj < 4; ++nj)
        acc[mi][nj] = wmma_bf16(af[mi], bfr[nj], acc[mi][nj]);
    __syncthreads();
  }

  const int halfsel = lane >> 4;
#pragma unroll
  for (int mi = 0; mi < 2; ++mi) {
#pragma unroll
    for (int nj = 0; nj < 4; ++nj) {
      const int n_g = nBase + wcol * 64 + nj * 16 + (lane & 15);
      const float bv = bias[n_g];
      const int m0 = mBase + wrow * 32 + mi * 16 + halfsel * 8;
      if (MODE == 0) {
#pragma unroll
        for (int i = 0; i < 8; ++i)
          outF[(size_t)(m0 + i) * N + n_g] = acc[mi][nj][i] + bv;
      } else {
        // V cache layout: [B,H,Dh,S]; 8 consecutive s -> one 16B store
        const int bidx = m0 >> 11, s0 = m0 & (SEQ - 1);
        const int h = n_g >> 6, d = n_g & 63;
        v8bf pv;
#pragma unroll
        for (int i = 0; i < 8; ++i) pv[i] = (__bf16)(acc[mi][nj][i] + bv);
        *reinterpret_cast<v8bf*>(
            outAttn + (((size_t)(bidx * HEADS + h) * DH + d) * SEQ) + s0) = pv;
      }
    }
  }
}

// ---------------------------------------------------------------------------
// 3) RMSNorm (over inner=2048) * scale, then RoPE; fp32 in -> bf16 [B,H,S,Dh]
// ---------------------------------------------------------------------------
__global__ __launch_bounds__(256) void normrope_kernel(
    const float* __restrict__ X, const float* __restrict__ scale,
    const float* __restrict__ cosT, const float* __restrict__ sinT,
    __bf16* __restrict__ outBf) {
  const int tok = blockIdx.x;
  const int t   = threadIdx.x;
  const float* row = X + (size_t)tok * DMODEL;
  const int c0 = t * 8;
  float4 x0 = reinterpret_cast<const float4*>(row + c0)[0];
  float4 x1 = reinterpret_cast<const float4*>(row + c0)[1];
  float ss = x0.x * x0.x + x0.y * x0.y + x0.z * x0.z + x0.w * x0.w +
             x1.x * x1.x + x1.y * x1.y + x1.z * x1.z + x1.w * x1.w;

  __shared__ float red[256];
  red[t] = ss;
  __syncthreads();
  for (int off = 128; off > 0; off >>= 1) {
    if (t < off) red[t] += red[t + off];
    __syncthreads();
  }
  const float rinv = rsqrtf(red[0] * (1.0f / DMODEL) + 1e-6f);

  const int s = tok & (SEQ - 1);
  const int b = tok >> 11;
  float4 s0 = reinterpret_cast<const float4*>(scale + c0)[0];
  float4 s1 = reinterpret_cast<const float4*>(scale + c0)[1];
  float xv[8] = {x0.x * s0.x, x0.y * s0.y, x0.z * s0.z, x0.w * s0.w,
                 x1.x * s1.x, x1.y * s1.y, x1.z * s1.z, x1.w * s1.w};
  const int h = c0 >> 6;
  const int dbase = c0 & 63;
  const float* cp = cosT + (size_t)s * DH;
  const float* sp = sinT + (size_t)s * DH;
  __bf16* dst = outBf + (((size_t)(b * HEADS + h) * SEQ + s) << 6) + dbase;
#pragma unroll
  for (int p = 0; p < 4; ++p) {
    const float xe = xv[2 * p] * rinv, xo = xv[2 * p + 1] * rinv;
    const float cs = cp[dbase + 2 * p], sn = sp[dbase + 2 * p];
    dst[2 * p]     = (__bf16)(xe * cs - xo * sn);
    dst[2 * p + 1] = (__bf16)(xo * cs + xe * sn);
  }
}

// ---------------------------------------------------------------------------
// 4) Flash attention: one block per (64-query tile, head, batch).
//    Q/K: [B,H,S,Dh], V: [B,H,Dh,S] (so P@V B-operand is transposed in LDS).
//    K/V tiles double-buffered through the Tensor Data Mover: tile i+1 is
//    issued before computing tile i; s_wait_tensorcnt(2) leaves the two
//    in-flight loads pending (TDM completes in order within a wave).
// ---------------------------------------------------------------------------
__global__ __launch_bounds__(256) void attn_kernel(
    const __bf16* __restrict__ qbf, const __bf16* __restrict__ kbf,
    const __bf16* __restrict__ vbf, __bf16* __restrict__ abf) {
  __shared__ alignas(16) __bf16 Qs[64][72];
  __shared__ alignas(16) __bf16 Ks[2][64][72];
  __shared__ alignas(16) __bf16 Vt[2][64][72];   // [dh][key]
  __shared__ alignas(16) __bf16 Pb[64][72];
  __shared__ float Sc[64][68];
  __shared__ float Oa[64][68];
  __shared__ float mrow[64], lrow[64], frow[64], red4[64][4];

  const int t = threadIdx.x;
  const int lane = t & 31;
  const int w = t >> 5;
  const int qb = blockIdx.x * 64;
  const int h  = blockIdx.y;
  const int b  = blockIdx.z;
  const size_t headOff = ((size_t)(b * HEADS + h)) * SEQ * DH;
  const __bf16* Qg = qbf + headOff;
  const __bf16* Kg = kbf + headOff;
  const __bf16* Vg = vbf + headOff;   // [Dh][S]

  const int lr = t >> 2, lc = (t & 3) * 16;

#if USE_TDM
  if (w == 0) {  // prologue: Q + first K/V tiles
    tdm_load_2d_bf16((unsigned)(uintptr_t)&Qs[0][0], Qg + (size_t)qb * DH,
                     DH, SEQ, DH, 64, 64);
    tdm_load_2d_bf16((unsigned)(uintptr_t)&Ks[0][0][0], Kg, DH, SEQ, DH, 64, 64);
    tdm_load_2d_bf16((unsigned)(uintptr_t)&Vt[0][0][0], Vg, SEQ, DH, SEQ, 64, 64);
  }
#else
  {
    const uint4* g = reinterpret_cast<const uint4*>(Qg + (size_t)(qb + lr) * DH + lc);
    uint4* d = reinterpret_cast<uint4*>(&Qs[lr][lc]);
    d[0] = g[0]; d[1] = g[1];
  }
#endif
  {  // init O accumulator + softmax state
    float* oz = &Oa[0][0];
    for (int i = t; i < 64 * 68; i += 256) oz[i] = 0.0f;
    if (t < 64) { mrow[t] = -1e30f; lrow[t] = 0.0f; }
  }
  __syncthreads();

  for (int it = 0; it < NKT; ++it) {
    const int kt  = it * 64;
    const int buf = it & 1;
#if USE_TDM
    if (w == 0) {
      if (it + 1 < NKT) {  // issue next tile into the other buffer
        tdm_load_2d_bf16((unsigned)(uintptr_t)&Ks[buf ^ 1][0][0],
                         Kg + (size_t)(kt + 64) * DH, DH, SEQ, DH, 64, 64);
        tdm_load_2d_bf16((unsigned)(uintptr_t)&Vt[buf ^ 1][0][0],
                         Vg + kt + 64, SEQ, DH, SEQ, 64, 64);
        __builtin_amdgcn_s_wait_tensorcnt(2);  // current buffer resident
      } else {
        __builtin_amdgcn_s_wait_tensorcnt(0);
      }
    }
    __syncthreads();
#else
    {
      const uint4* gk = reinterpret_cast<const uint4*>(Kg + (size_t)(kt + lr) * DH + lc);
      uint4* dk = reinterpret_cast<uint4*>(&Ks[buf][lr][lc]);
      dk[0] = gk[0]; dk[1] = gk[1];
      const uint4* gv = reinterpret_cast<const uint4*>(Vg + (size_t)lr * SEQ + kt + lc);
      uint4* dv = reinterpret_cast<uint4*>(&Vt[buf][lr][lc]);
      dv[0] = gv[0]; dv[1] = gv[1];
    }
    __syncthreads();
#endif

    // scores = (Q @ K^T) * Dh^-0.5 ; 16 subtiles of 16x16, 2 per wave
#pragma unroll
    for (int ssub = 0; ssub < 2; ++ssub) {
      const int idx = w + ssub * 8;
      const int ti = idx >> 2, tj = idx & 3;
      v8f acc = {};
#pragma unroll
      for (int ks = 0; ks < 2; ++ks) {
        v16bf a  = load_a_frag(&Qs[0][0], 72, ti * 16, ks * 32, lane);
        v16bf bm = load_bt_frag(&Ks[buf][0][0], 72, ks * 32, tj * 16, lane);
        acc = wmma_bf16(a, bm, acc);
      }
      const int col = tj * 16 + (lane & 15);
      const int rb  = ti * 16 + ((lane >> 4) << 3);
#pragma unroll
      for (int i = 0; i < 8; ++i) Sc[rb + i][col] = acc[i] * 0.125f;
    }
    __syncthreads();

    // online softmax: 4 threads per row, 16 cols each (vectorized)
    {
      const int r = lr, part = t & 3, cc = (t & 3) * 16;
      float4 sv[4];
#pragma unroll
      for (int q = 0; q < 4; ++q)
        sv[q] = *reinterpret_cast<const float4*>(&Sc[r][cc + 4 * q]);
      float mx = -1e30f;
#pragma unroll
      for (int q = 0; q < 4; ++q)
        mx = fmaxf(mx, fmaxf(fmaxf(sv[q].x, sv[q].y), fmaxf(sv[q].z, sv[q].w)));
      red4[r][part] = mx;
      __syncthreads();
      if (part == 0) {
        float mnew = fmaxf(fmaxf(red4[r][0], red4[r][1]),
                           fmaxf(red4[r][2], red4[r][3]));
        mnew = fmaxf(mnew, mrow[r]);
        frow[r] = __expf(mrow[r] - mnew);
        mrow[r] = mnew;
      }
      __syncthreads();
      const float mnew = mrow[r];
      const float f = frow[r];
      float p[16];
#pragma unroll
      for (int q = 0; q < 4; ++q) {
        p[4 * q + 0] = __expf(sv[q].x - mnew);
        p[4 * q + 1] = __expf(sv[q].y - mnew);
        p[4 * q + 2] = __expf(sv[q].z - mnew);
        p[4 * q + 3] = __expf(sv[q].w - mnew);
      }
      float sum = 0.0f;
      v8bf pb0, pb1;
#pragma unroll
      for (int j = 0; j < 8; ++j) { sum += p[j]; pb0[j] = (__bf16)p[j]; }
#pragma unroll
      for (int j = 0; j < 8; ++j) { sum += p[8 + j]; pb1[j] = (__bf16)p[8 + j]; }
      *reinterpret_cast<v8bf*>(&Pb[r][cc])     = pb0;
      *reinterpret_cast<v8bf*>(&Pb[r][cc + 8]) = pb1;
      red4[r][part] = sum;
      float4* oa4 = reinterpret_cast<float4*>(&Oa[r][cc]);
#pragma unroll
      for (int q = 0; q < 4; ++q) {
        float4 ov = oa4[q];
        ov.x *= f; ov.y *= f; ov.z *= f; ov.w *= f;
        oa4[q] = ov;
      }
      __syncthreads();
      if (part == 0)
        lrow[r] = lrow[r] * f + red4[r][0] + red4[r][1] + red4[r][2] + red4[r][3];
    }
    __syncthreads();

    // O += P @ V   (B-operand from transposed V tile -> contiguous fragments)
#pragma unroll
    for (int ssub = 0; ssub < 2; ++ssub) {
      const int idx = w + ssub * 8;
      const int ti = idx >> 2, tj = idx & 3;
      const int col = tj * 16 + (lane & 15);
      const int rb  = ti * 16 + ((lane >> 4) << 3);
      v8f acc;
#pragma unroll
      for (int i = 0; i < 8; ++i) acc[i] = Oa[rb + i][col];
#pragma unroll
      for (int ks = 0; ks < 2; ++ks) {
        v16bf a  = load_a_frag(&Pb[0][0], 72, ti * 16, ks * 32, lane);
        v16bf bm = load_bt_frag(&Vt[buf][0][0], 72, ks * 32, tj * 16, lane);
        acc = wmma_bf16(a, bm, acc);
      }
#pragma unroll
      for (int i = 0; i < 8; ++i) Oa[rb + i][col] = acc[i];
    }
    __syncthreads();
  }

  // epilogue: O /= l, write bf16 into [B,S,H*Dh]
  {
    const int r = lr, cc = lc;
    const float linv = 1.0f / lrow[r];
    const int s = qb + r;
    const float4* oa4 = reinterpret_cast<const float4*>(&Oa[r][cc]);
    v8bf o0, o1;
#pragma unroll
    for (int q = 0; q < 2; ++q) {
      float4 v = oa4[q];
      o0[4 * q + 0] = (__bf16)(v.x * linv); o0[4 * q + 1] = (__bf16)(v.y * linv);
      o0[4 * q + 2] = (__bf16)(v.z * linv); o0[4 * q + 3] = (__bf16)(v.w * linv);
    }
#pragma unroll
    for (int q = 0; q < 2; ++q) {
      float4 v = oa4[2 + q];
      o1[4 * q + 0] = (__bf16)(v.x * linv); o1[4 * q + 1] = (__bf16)(v.y * linv);
      o1[4 * q + 2] = (__bf16)(v.z * linv); o1[4 * q + 3] = (__bf16)(v.w * linv);
    }
    __bf16* dst = abf + ((size_t)(b * SEQ + s)) * (HEADS * DH) + h * DH + cc;
    *reinterpret_cast<v8bf*>(dst)     = o0;
    *reinterpret_cast<v8bf*>(dst + 8) = o1;
  }
}

// ---------------------------------------------------------------------------
// Host launcher
// ---------------------------------------------------------------------------
extern "C" void kernel_launch(void* const* d_in, const int* in_sizes, int n_in,
                              void* d_out, int out_size, void* d_ws, size_t ws_size,
                              hipStream_t stream) {
  (void)in_sizes; (void)n_in; (void)out_size; (void)ws_size;
  const float* hidden = (const float*)d_in[0];
  const float* cosT   = (const float*)d_in[1];
  const float* sinT   = (const float*)d_in[2];
  const float* Wq = (const float*)d_in[3];
  const float* bq = (const float*)d_in[4];
  const float* Wk = (const float*)d_in[5];
  const float* bk = (const float*)d_in[6];
  const float* Wv = (const float*)d_in[7];
  const float* bv = (const float*)d_in[8];
  const float* qs  = (const float*)d_in[9];
  const float* ksc = (const float*)d_in[10];
  const float* Wo = (const float*)d_in[11];
  const float* bo = (const float*)d_in[12];
  float* out = (float*)d_out;

  char* ws = (char*)d_ws;
  const size_t SZ_BF  = (size_t)NTOK * DMODEL * sizeof(__bf16);  // 16 MB
  const size_t SZ_F32 = (size_t)NTOK * DMODEL * sizeof(float);   // 32 MB
  __bf16* Xbf = (__bf16*)(ws);
  float*  q32 = (float*)(ws + SZ_BF);
  float*  k32 = (float*)(ws + SZ_BF + SZ_F32);
  __bf16* qbf = (__bf16*)(ws + SZ_BF + 2 * SZ_F32);
  __bf16* kbf = (__bf16*)(ws + 2 * SZ_BF + 2 * SZ_F32);
  __bf16* vbf = (__bf16*)(ws + 3 * SZ_BF + 2 * SZ_F32);
  __bf16* abf = (__bf16*)(ws + SZ_BF);  // aliases q32 (dead after normrope)

  dim3 blk(256);
  cvt_kernel<<<dim3((NTOK * DMODEL / 4 + 255) / 256), blk, 0, stream>>>(
      hidden, Xbf, NTOK * DMODEL / 4);

  dim3 ggrid(DMODEL / 128, NTOK / 128);
  gemm_bf16_kernel<0><<<ggrid, blk, 0, stream>>>(Xbf, Wq, bq, q32, nullptr,
                                                 NTOK, DMODEL, DMODEL);
  gemm_bf16_kernel<0><<<ggrid, blk, 0, stream>>>(Xbf, Wk, bk, k32, nullptr,
                                                 NTOK, DMODEL, DMODEL);
  gemm_bf16_kernel<1><<<ggrid, blk, 0, stream>>>(Xbf, Wv, bv, nullptr, vbf,
                                                 NTOK, DMODEL, DMODEL);

  normrope_kernel<<<dim3(NTOK), blk, 0, stream>>>(q32, qs, cosT, sinT, qbf);
  normrope_kernel<<<dim3(NTOK), blk, 0, stream>>>(k32, ksc, cosT, sinT, kbf);

  attn_kernel<<<dim3(SEQ / 64, HEADS, 2), blk, 0, stream>>>(qbf, kbf, vbf, abf);

  gemm_bf16_kernel<0><<<ggrid, blk, 0, stream>>>(abf, Wo, bo, out, nullptr,
                                                 NTOK, DMODEL, DMODEL);
}
